// TropicalLeNet1_50379966382852
// MI455X (gfx1250) — matrix operand order repaired and checked by reading, not compiled, not run on hardware
//
#include <hip/hip_runtime.h>
#include <hip/hip_bf16.h>
#include <hip/hip_fp16.h>

typedef __attribute__((ext_vector_type(16))) _Float16 v16h;
typedef __attribute__((ext_vector_type(8)))  _Float16 v8h;
typedef __attribute__((ext_vector_type(8)))  float    v8f;

// ---- problem constants ----
#define NB     64      // batch
#define CIN    3
#define LIN    8192
#define O1     6
#define K1     80
#define LP1    4058    // pooled length after conv1 (8117 -> 4058)
#define O2     16
#define K2     3
#define LP2    2028    // pooled length after conv2 (4056 -> 2028)
#define NF     32448   // 16*2028 flattened features
#define NFPAD  128     // fc1 N padded to 128
#define H1     120
#define H2     84
#define NOUT   10
#define KSLICE 6       // fc1 K-split (1014 wmma steps = 6*169)
#define KSTEPS 169

// ============================================================
// Kernel 1: min-plus conv1 (pad=2, K=80) fused with avg-pool/2
// y[b,o,tp] = 0.5*(c(2tp)+c(2tp+1)), c(t)=sum_ch min_k (xpad+w)
// ============================================================
__global__ void conv1_minplus_pool(const float* __restrict__ x,
                                   const float* __restrict__ w1,
                                   float* __restrict__ x1)
{
    const int tid = threadIdx.x;
    const int tp0 = blockIdx.x * 256;
    const int o   = blockIdx.y;
    const int b   = blockIdx.z;

    __shared__ float xl[3 * 592];
    __shared__ float wl[3 * 80];

    for (int i = tid; i < CIN * K1; i += 256) wl[i] = w1[o * CIN * K1 + i];

    const int base = 2 * tp0;   // xpad index of xl[ch][0]
    for (int ch = 0; ch < CIN; ++ch) {
        const float* xr = x + ((size_t)b * CIN + ch) * LIN;
        for (int i = tid; i < 592; i += 256) {
            int gi = base + i - 2;                   // zero padding of 2
            xl[ch * 592 + i] = (gi >= 0 && gi < LIN) ? xr[gi] : 0.0f;
        }
    }
    __syncthreads();

    const int tp = tp0 + tid;
    if (tp >= LP1) return;

    float s0 = 0.f, s1 = 0.f;
    #pragma unroll
    for (int ch = 0; ch < CIN; ++ch) {
        const float* xb = &xl[ch * 592 + 2 * tid];
        const float* wb = &wl[ch * K1];
        float m0 = __builtin_inff(), m1 = __builtin_inff();
        for (int k = 0; k < K1; ++k) {
            float wv = wb[k];
            m0 = fminf(m0, xb[k]     + wv);
            m1 = fminf(m1, xb[k + 1] + wv);
        }
        s0 += m0; s1 += m1;
    }
    x1[((size_t)b * O1 + o) * LP1 + tp] = 0.5f * (s0 + s1);
}

// ============================================================
// Kernel 2: max-plus conv2 (K=3) fused with avg-pool/2,
// output stored as f16 in flattened [b][o*2028+tp] layout
// ============================================================
__global__ void conv2_maxplus_pool(const float* __restrict__ x1,
                                   const float* __restrict__ w2,
                                   _Float16* __restrict__ x2h)
{
    const int tid = threadIdx.x;
    const int tp0 = blockIdx.x * 256;
    const int o   = blockIdx.y;
    const int b   = blockIdx.z;

    __shared__ float xl[6 * 514];
    __shared__ float wl[O1 * K2];

    if (tid < O1 * K2) wl[tid] = w2[o * O1 * K2 + tid];

    const int base = 2 * tp0;
    for (int ch = 0; ch < O1; ++ch) {
        const float* xr = x1 + ((size_t)b * O1 + ch) * LP1;
        for (int i = tid; i < 514; i += 256) {
            int gi = base + i;
            xl[ch * 514 + i] = (gi < LP1) ? xr[gi] : 0.0f;
        }
    }
    __syncthreads();

    const int tp = tp0 + tid;
    if (tp >= LP2) return;

    float s0 = 0.f, s1 = 0.f;
    #pragma unroll
    for (int ch = 0; ch < O1; ++ch) {
        const float* xb = &xl[ch * 514 + 2 * tid];
        const float* wb = &wl[ch * K2];
        float m0 = -__builtin_inff(), m1 = -__builtin_inff();
        #pragma unroll
        for (int k = 0; k < K2; ++k) {
            float wv = wb[k];
            m0 = fmaxf(m0, xb[k]     + wv);
            m1 = fmaxf(m1, xb[k + 1] + wv);
        }
        s0 += m0; s1 += m1;
    }
    x2h[(size_t)b * NF + o * LP2 + tp] = (_Float16)(0.5f * (s0 + s1));
}

// ============================================================
// Kernel 3: fc1_w fp32 [120][32448] -> f16 [128][32448] (rows
// 120..127 zero so padded N-tiles are harmless)
// ============================================================
__global__ void convert_fc1w(const float* __restrict__ fc1_w,
                             _Float16* __restrict__ wh)
{
    int idx = blockIdx.x * 256 + threadIdx.x;
    if (idx >= NFPAD * NF) return;
    int n = idx / NF;
    wh[idx] = (_Float16)((n < H1) ? fc1_w[idx] : 0.0f);
}

// ============================================================
// Kernel 4: fc1 GEMM via v_wmma_f32_16x16x32_f16.
// grid=(4 M-tiles, 6 K-slices), block=256 (8 waves -> N-tiles).
// Partial sums -> part[ks][64][128] (deterministic reduce later).
// ============================================================
__global__ void fc1_wmma_partial(const _Float16* __restrict__ A,   // [64][NF]
                                 const _Float16* __restrict__ Bw,  // [128][NF]
                                 float* __restrict__ part)         // [6][64][128]
{
    const int lane = threadIdx.x & 31;
    const int wave = threadIdx.x >> 5;    // N-tile 0..7
    const int mt   = blockIdx.x;          // M-tile 0..3
    const int ks   = blockIdx.y;          // K-slice 0..5

    const int  half_lane = lane & 15;
    const bool hi        = lane >= 16;

    // A layout (matches B): lanes 0-15 hold K 0-15, lanes 16-31 K 16-31
    const _Float16* ap = A  + (size_t)(mt * 16 + half_lane)   * NF + (hi ? 16 : 0);
    // B layout (same contiguous k order as A): lanes 0-15 hold K 0-15, lanes 16-31 K 16-31 (column-major per N)
    const _Float16* bp = Bw + (size_t)(wave * 16 + half_lane) * NF + (hi ? 16 : 0);

    const int kbase = ks * KSTEPS * 32;

    v8f c = {};
    for (int s = 0; s < KSTEPS; ++s) {
        const int k0 = kbase + s * 32;
        v8h alo = *reinterpret_cast<const v8h*>(ap + k0);        // 16B load
        v8h ahi = *reinterpret_cast<const v8h*>(ap + k0 + 8);    // 16B load
        v16h a  = __builtin_shufflevector(alo, ahi,
                    0,1,2,3,4,5,6,7,8,9,10,11,12,13,14,15);
        v16h bf = *reinterpret_cast<const v16h*>(bp + k0);       // 32B load
        c = __builtin_amdgcn_wmma_f32_16x16x32_f16(
                /*neg_a=*/false, a, /*neg_b=*/false, bf,
                /*c_mod=*/(short)0, c, /*reuse_a=*/false, /*reuse_b=*/false);
    }

    // C/D layout: lanes 0-15 N=0-15; VGPR r holds M=r (lo lanes) / M=r+8 (hi lanes)
    #pragma unroll
    for (int r = 0; r < 8; ++r) {
        int row = mt * 16 + r + (hi ? 8 : 0);
        part[((size_t)ks * NB + row) * NFPAD + wave * 16 + half_lane] = c[r];
    }
}

__global__ void fc1_reduce_relu(const float* __restrict__ part,
                                const float* __restrict__ bias,
                                float* __restrict__ Y1)           // [64][120]
{
    int t = blockIdx.x * 256 + threadIdx.x;
    if (t >= NB * H1) return;
    int row = t / H1, col = t % H1;
    float s = bias[col];
    #pragma unroll
    for (int ks = 0; ks < KSLICE; ++ks)
        s += part[((size_t)ks * NB + row) * NFPAD + col];
    Y1[t] = s > 0.f ? s : 0.f;
}

// ============================================================
// Kernels 5/6: tiny FC layers in exact fp32
// ============================================================
__global__ void fc2_relu(const float* __restrict__ Y1, const float* __restrict__ W,
                         const float* __restrict__ bias, float* __restrict__ Y2)
{
    int t = blockIdx.x * 256 + threadIdx.x;
    if (t >= NB * H2) return;
    int b = t / H2, n = t % H2;
    const float* xr = Y1 + b * H1;
    const float* wr = W  + n * H1;
    float s = bias[n];
    for (int k = 0; k < H1; ++k) s += xr[k] * wr[k];
    Y2[t] = s > 0.f ? s : 0.f;
}

__global__ void fc3_out(const float* __restrict__ Y2, const float* __restrict__ W,
                        const float* __restrict__ bias, float* __restrict__ out)
{
    int t = blockIdx.x * 256 + threadIdx.x;
    if (t >= NB * NOUT) return;
    int b = t / NOUT, n = t % NOUT;
    const float* xr = Y2 + b * H2;
    const float* wr = W  + n * H2;
    float s = bias[n];
    for (int k = 0; k < H2; ++k) s += xr[k] * wr[k];
    out[t] = s;
}

// ============================================================
// launch
// ============================================================
extern "C" void kernel_launch(void* const* d_in, const int* in_sizes, int n_in,
                              void* d_out, int out_size, void* d_ws, size_t ws_size,
                              hipStream_t stream)
{
    const float* x     = (const float*)d_in[0];
    const float* w1    = (const float*)d_in[1];
    const float* w2    = (const float*)d_in[2];
    const float* fc1_w = (const float*)d_in[3];
    const float* fc1_b = (const float*)d_in[4];
    const float* fc2_w = (const float*)d_in[5];
    const float* fc2_b = (const float*)d_in[6];
    const float* fc3_w = (const float*)d_in[7];
    const float* fc3_b = (const float*)d_in[8];
    float* out = (float*)d_out;

    // workspace layout (all offsets 256B aligned)
    char* ws = (char*)d_ws;
    float*    x1     = (float*)   (ws + 0);           //  6,233,088 B  [64][6][4058] f32
    _Float16* x2h    = (_Float16*)(ws + 6233088);     //  4,153,344 B  [64][32448]  f16
    _Float16* fc1_wh = (_Float16*)(ws + 10386432);    //  8,306,688 B  [128][32448] f16
    float*    part   = (float*)   (ws + 18693120);    //    196,608 B  [6][64][128] f32
    float*    y1     = (float*)   (ws + 18889728);    //     30,720 B  [64][120]    f32
    float*    y2     = (float*)   (ws + 18920448);    //     21,504 B  [64][84]     f32
    // total ~18.95 MB

    conv1_minplus_pool<<<dim3(16, O1, NB), 256, 0, stream>>>(x, w1, x1);
    conv2_maxplus_pool<<<dim3(8, O2, NB), 256, 0, stream>>>(x1, w2, x2h);
    convert_fc1w<<<(NFPAD * NF) / 256, 256, 0, stream>>>(fc1_w, fc1_wh);
    fc1_wmma_partial<<<dim3(4, KSLICE), 256, 0, stream>>>(x2h, fc1_wh, part);
    fc1_reduce_relu<<<(NB * H1 + 255) / 256, 256, 0, stream>>>(part, fc1_b, y1);
    fc2_relu<<<(NB * H2 + 255) / 256, 256, 0, stream>>>(y1, fc2_w, fc2_b, y2);
    fc3_out<<<(NB * NOUT + 255) / 256, 256, 0, stream>>>(y2, fc3_w, fc3_b, out);
}
